// CrossAttention_55439437857301
// MI455X (gfx1250) — compile-verified, matrix operand order checked by reading, not run on hardware
//
#include <hip/hip_runtime.h>
#include <hip/hip_bf16.h>
#include <math.h>

// Problem dims (fixed by reference)
#define BB 4
#define CC 256
#define KC8 32
#define NN 4096
#define MT 4            // m-subtiles (16 cols each) per passB block => 64 columns

typedef __attribute__((ext_vector_type(16))) __bf16 v16bf;
typedef __attribute__((ext_vector_type(8)))  float  v8f;

union Frag { v16bf v; uint4 q[2]; };

__device__ __forceinline__ unsigned short f2bf(float x) {
    unsigned u = __float_as_uint(x);
    u += 0x7fffu + ((u >> 16) & 1u);   // round-to-nearest-even
    return (unsigned short)(u >> 16);
}
__device__ __forceinline__ unsigned pk2(float a, float b) {
    return (unsigned)f2bf(a) | ((unsigned)f2bf(b) << 16);
}

// Load a 16x32 bf16 WMMA operand fragment from row-major [row][k] storage.
// ISA 16-bit A layout: lanes 0-15 hold rows 0-15 with k in {0..7, 16..23},
// lanes 16-31 hold rows 0-15 with k in {8..15, 24..31}; k pairs packed per VGPR.
// => per lane: two contiguous 16-byte loads.
__device__ __forceinline__ v16bf load_frag(const unsigned short* __restrict__ base,
                                           int r0, int ld, int k0) {
    int lane = threadIdx.x & 31;
    const unsigned short* p = base + (size_t)(r0 + (lane & 15)) * ld + k0 + ((lane >> 4) << 3);
    Frag f;
    f.q[0] = *(const uint4*)(p);
    f.q[1] = *(const uint4*)(p + 16);
    return f.v;
}

__device__ __forceinline__ v8f wmma_bf16(v16bf a, v16bf b, v8f c) {
    return __builtin_amdgcn_wmma_f32_16x16x32_bf16(false, a, false, b, (short)0, c, false, false);
}

// ---------------------------------------------------------------------------
// Kernel 1: transpose f32 [C][N] -> bf16 [N][C] per batch (LDS tiled).
__global__ void k_transpose_bf16(const float* __restrict__ src, unsigned short* __restrict__ dst) {
    __shared__ float tile[32][33];
    int b = blockIdx.z;
    const float* s = src + (size_t)b * CC * NN;
    unsigned short* d = dst + (size_t)b * NN * CC;
    int n0 = blockIdx.x * 32, c0 = blockIdx.y * 32;
#pragma unroll
    for (int i = 0; i < 4; ++i) {
        int c = threadIdx.y + i * 8;
        tile[c][threadIdx.x] = s[(size_t)(c0 + c) * NN + n0 + threadIdx.x];
    }
    __syncthreads();
#pragma unroll
    for (int i = 0; i < 4; ++i) {
        int n = threadIdx.y + i * 8;
        d[(size_t)(n0 + n) * CC + c0 + threadIdx.x] = f2bf(tile[threadIdx.x][n]);
    }
}

// ---------------------------------------------------------------------------
// Kernel 2: pack Wf(32x256) | Wg(32x256) | Wh(256x256) -> bf16 [320][256], biases -> f32[320]
__global__ void k_pack_w(const float* __restrict__ Wf, const float* __restrict__ bf_,
                         const float* __restrict__ Wg, const float* __restrict__ bg,
                         const float* __restrict__ Wh, const float* __restrict__ bh,
                         unsigned short* __restrict__ Wp, float* __restrict__ bp) {
    int o = blockIdx.x, c = threadIdx.x;
    const float* Ws; const float* bs; int oo;
    if (o < 32)       { Ws = Wf; bs = bf_; oo = o; }
    else if (o < 64)  { Ws = Wg; bs = bg;  oo = o - 32; }
    else              { Ws = Wh; bs = bh;  oo = o - 64; }
    Wp[o * CC + c] = f2bf(Ws[oo * CC + c]);
    if (c == 0) bp[o] = bs[oo];
}

// ---------------------------------------------------------------------------
// Kernel 3: projections. P[320][N] = Wp(320x256) @ X(256xN) + b, per batch.
// Outputs: FT,GT as bf16 [N][32] (transposed), H as bf16 [C][N].
// One wave per 16x16 output tile; 8 waves per block; grid = (20*256/8, B).
__global__ void k_proj(const unsigned short* __restrict__ XT,   // [N][C] bf16
                       const unsigned short* __restrict__ Wp,   // [320][C] bf16
                       const float* __restrict__ bp,            // [320]
                       unsigned short* __restrict__ FT,         // [N][32]
                       unsigned short* __restrict__ GT,         // [N][32]
                       unsigned short* __restrict__ H) {        // [C][N]
    int b = blockIdx.y;
    const unsigned short* xt = XT + (size_t)b * NN * CC;
    int wave = blockIdx.x * (blockDim.x >> 5) + (threadIdx.x >> 5);
    int lane = threadIdx.x & 31;
    int ot = wave / (NN / 16);
    int nt = wave % (NN / 16);
    v8f acc = {};
#pragma unroll
    for (int kk = 0; kk < CC; kk += 32) {
        v16bf a = load_frag(Wp, ot * 16, CC, kk);   // A: W rows (o x k=c)
        v16bf x = load_frag(xt, nt * 16, CC, kk);   // B: X cols (k=c x n)
        acc = wmma_bf16(a, x, acc);
    }
    int n = nt * 16 + (lane & 15);
    int obase = ot * 16 + ((lane >> 4) << 3);
    float4 b0 = *(const float4*)(bp + obase);
    float4 b1 = *(const float4*)(bp + obase + 4);
    float v[8];
    v[0] = acc[0] + b0.x; v[1] = acc[1] + b0.y; v[2] = acc[2] + b0.z; v[3] = acc[3] + b0.w;
    v[4] = acc[4] + b1.x; v[5] = acc[5] + b1.y; v[6] = acc[6] + b1.z; v[7] = acc[7] + b1.w;
    if (ot < 4) {  // f (ot 0,1) or g (ot 2,3): store transposed [n][32]
        unsigned short* dst = (ot < 2 ? FT : GT) + (size_t)b * NN * KC8;
        int oo = (ot & 1) * 16 + ((lane >> 4) << 3);
        uint4 pk;
        pk.x = pk2(v[0], v[1]); pk.y = pk2(v[2], v[3]);
        pk.z = pk2(v[4], v[5]); pk.w = pk2(v[6], v[7]);
        *(uint4*)(dst + (size_t)n * KC8 + oo) = pk;
    } else {       // h: [C][N]
        unsigned short* dst = H + (size_t)b * CC * NN;
#pragma unroll
        for (int r = 0; r < 8; ++r)
            dst[(size_t)(obase - 64 + r) * NN + n] = f2bf(v[r]);
    }
}

// ---------------------------------------------------------------------------
// Kernel 4: pass A — L[n] = max_m S[n,m] + log(sum_m exp(S[n,m]-max)),
// S = fq^T @ gk, K = 32 (one WMMA per 16x16 tile).
// One wave per 32-row n-block (2 tiles) so every gk fragment feeds 2 WMMAs.
__global__ void k_passA(const unsigned short* __restrict__ FTq,  // [N][32]
                        const unsigned short* __restrict__ GTk,  // [N][32]
                        float* __restrict__ L) {                 // [B][N]
    int wid = blockIdx.x * 8 + (threadIdx.x >> 5);
    int lane = threadIdx.x & 31;
    int b = wid >> 7;            // / (N/32)
    int nt2 = wid & 127;         // % (N/32)
    int n0 = nt2 * 32;
    const unsigned short* fq = FTq + (size_t)b * NN * KC8;
    const unsigned short* gk = GTk + (size_t)b * NN * KC8;
    v16bf a0 = load_frag(fq, n0, KC8, 0);
    v16bf a1 = load_frag(fq, n0 + 16, KC8, 0);
    float mx0[8], sm0[8], mx1[8], sm1[8];
#pragma unroll
    for (int r = 0; r < 8; ++r) {
        mx0[r] = -3.0e38f; sm0[r] = 0.f;
        mx1[r] = -3.0e38f; sm1[r] = 0.f;
    }
    for (int mt = 0; mt < NN / 16; ++mt) {
        v16bf bb = load_frag(gk, mt * 16, KC8, 0);
        v8f s0 = {}, s1 = {};
        s0 = wmma_bf16(a0, bb, s0);
        s1 = wmma_bf16(a1, bb, s1);
#pragma unroll
        for (int r = 0; r < 8; ++r) {
            float v0 = s0[r];
            float nm0 = fmaxf(mx0[r], v0);
            sm0[r] = sm0[r] * __expf(mx0[r] - nm0) + __expf(v0 - nm0);
            mx0[r] = nm0;
            float v1 = s1[r];
            float nm1 = fmaxf(mx1[r], v1);
            sm1[r] = sm1[r] * __expf(mx1[r] - nm1) + __expf(v1 - nm1);
            mx1[r] = nm1;
        }
    }
    // combine across the 16 lanes holding the same rows (xor-reduce, width 16)
#pragma unroll
    for (int d = 1; d < 16; d <<= 1) {
#pragma unroll
        for (int r = 0; r < 8; ++r) {
            float om = __shfl_xor(mx0[r], d, 16);
            float os = __shfl_xor(sm0[r], d, 16);
            float nm = fmaxf(mx0[r], om);
            sm0[r] = sm0[r] * __expf(mx0[r] - nm) + os * __expf(om - nm);
            mx0[r] = nm;
            om = __shfl_xor(mx1[r], d, 16);
            os = __shfl_xor(sm1[r], d, 16);
            nm = fmaxf(mx1[r], om);
            sm1[r] = sm1[r] * __expf(mx1[r] - nm) + os * __expf(om - nm);
            mx1[r] = nm;
        }
    }
    if ((lane & 15) == 0) {
        float* Lb = L + (size_t)b * NN + n0 + ((lane >> 4) << 3);
        float4 o0, o1;
        o0.x = mx0[0] + __logf(sm0[0]); o0.y = mx0[1] + __logf(sm0[1]);
        o0.z = mx0[2] + __logf(sm0[2]); o0.w = mx0[3] + __logf(sm0[3]);
        o1.x = mx0[4] + __logf(sm0[4]); o1.y = mx0[5] + __logf(sm0[5]);
        o1.z = mx0[6] + __logf(sm0[6]); o1.w = mx0[7] + __logf(sm0[7]);
        *(float4*)(Lb) = o0;
        *(float4*)(Lb + 4) = o1;
        float4 p0, p1;
        p0.x = mx1[0] + __logf(sm1[0]); p0.y = mx1[1] + __logf(sm1[1]);
        p0.z = mx1[2] + __logf(sm1[2]); p0.w = mx1[3] + __logf(sm1[3]);
        p1.x = mx1[4] + __logf(sm1[4]); p1.y = mx1[5] + __logf(sm1[5]);
        p1.z = mx1[6] + __logf(sm1[6]); p1.w = mx1[7] + __logf(sm1[7]);
        *(float4*)(Lb + 16) = p0;
        *(float4*)(Lb + 20) = p1;
    }
}

// ---------------------------------------------------------------------------
// Kernel 5: pass B — out[c,m] = gamma * sum_n h[c,n] * exp(S[n,m]-L[n]) + resid[c,m].
// Block: 64 m-columns (4 subtiles), 8 waves covering 256 channels (2 c-tiles each).
// Each wave computes P for one 32-row n-chunk x 64 m-cols per super-step into LDS
// (this also converts C-layout -> B-operand layout), then all waves run PV WMMAs:
// per chunk, 2 h-fragment loads feed 8 WMMAs (4x h reuse over m-subtiles).
__global__ void k_passB(const unsigned short* __restrict__ FTq,  // [N][32]
                        const unsigned short* __restrict__ GTk,  // [N][32]
                        const unsigned short* __restrict__ Hsrc, // [C][N]
                        const float* __restrict__ L,             // [B][N]
                        const float* __restrict__ resid,         // f32 [C][N]
                        const float* __restrict__ gamma,
                        float* __restrict__ out) {               // f32 [C][N]
    __shared__ unsigned short Pl[8][16 * MT][32];   // [chunk][m][n_local] bf16, 32 KB
    int b = blockIdx.y;
    int m0 = blockIdx.x * (16 * MT);
    int w = threadIdx.x >> 5;
    int lane = threadIdx.x & 31;
    const unsigned short* fq = FTq + (size_t)b * NN * KC8;
    const unsigned short* gk = GTk + (size_t)b * NN * KC8;
    const unsigned short* hh = Hsrc + (size_t)b * CC * NN;
    const float* Lb = L + (size_t)b * NN;
    v16bf bg[MT];                               // score B operands, fixed per block
#pragma unroll
    for (int j = 0; j < MT; ++j) bg[j] = load_frag(gk, m0 + 16 * j, KC8, 0);
    v8f acc0[MT], acc1[MT];
#pragma unroll
    for (int j = 0; j < MT; ++j) { acc0[j] = {}; acc1[j] = {}; }
    int c0 = w * 32;
    int nb = (lane >> 4) << 3;
    for (int sup = 0; sup < NN / 256; ++sup) {
        int n0 = (sup * 8 + w) * 32;
        // --- scores + softmax numerator for this wave's 32-row n-chunk ---
        v16bf a0 = load_frag(fq, n0, KC8, 0);
        v16bf a1 = load_frag(fq, n0 + 16, KC8, 0);
        float4 l0a = *(const float4*)(Lb + n0 + nb);
        float4 l0b = *(const float4*)(Lb + n0 + nb + 4);
        float4 l1a = *(const float4*)(Lb + n0 + 16 + nb);
        float4 l1b = *(const float4*)(Lb + n0 + 16 + nb + 4);
#pragma unroll
        for (int j = 0; j < MT; ++j) {
            v8f s0 = {}, s1 = {};
            s0 = wmma_bf16(a0, bg[j], s0);
            s1 = wmma_bf16(a1, bg[j], s1);
            uint4 pk0, pk1;
            pk0.x = pk2(__expf(s0[0] - l0a.x), __expf(s0[1] - l0a.y));
            pk0.y = pk2(__expf(s0[2] - l0a.z), __expf(s0[3] - l0a.w));
            pk0.z = pk2(__expf(s0[4] - l0b.x), __expf(s0[5] - l0b.y));
            pk0.w = pk2(__expf(s0[6] - l0b.z), __expf(s0[7] - l0b.w));
            pk1.x = pk2(__expf(s1[0] - l1a.x), __expf(s1[1] - l1a.y));
            pk1.y = pk2(__expf(s1[2] - l1a.z), __expf(s1[3] - l1a.w));
            pk1.z = pk2(__expf(s1[4] - l1b.x), __expf(s1[5] - l1b.y));
            pk1.w = pk2(__expf(s1[6] - l1b.z), __expf(s1[7] - l1b.w));
            *(uint4*)&Pl[w][16 * j + (lane & 15)][nb] = pk0;        // [m][n] row-major
            *(uint4*)&Pl[w][16 * j + (lane & 15)][16 + nb] = pk1;
        }
        __syncthreads();
        // --- PV: accumulate over the 8 chunks of this super-step ---
#pragma unroll
        for (int ch = 0; ch < 8; ++ch) {
            int nk = sup * 256 + ch * 32;
            v16bf h0 = load_frag(hh, c0, NN, nk);
            v16bf h1 = load_frag(hh, c0 + 16, NN, nk);
#pragma unroll
            for (int j = 0; j < MT; ++j) {
                Frag fb;
                fb.q[0] = *(const uint4*)&Pl[ch][16 * j + (lane & 15)][nb];
                fb.q[1] = *(const uint4*)&Pl[ch][16 * j + (lane & 15)][nb + 16];
                acc0[j] = wmma_bf16(h0, fb.v, acc0[j]);
                acc1[j] = wmma_bf16(h1, fb.v, acc1[j]);
            }
        }
        __syncthreads();
    }
    // epilogue: gamma * sa + residual
    float g = gamma[0];
    const float* rb = resid + (size_t)b * CC * NN;
    float* ob = out + (size_t)b * CC * NN;
    int cb = c0 + nb;
#pragma unroll
    for (int j = 0; j < MT; ++j) {
        int m = m0 + 16 * j + (lane & 15);
#pragma unroll
        for (int r = 0; r < 8; ++r) {
            ob[(size_t)(cb + r) * NN + m] = g * acc0[j][r] + rb[(size_t)(cb + r) * NN + m];
            ob[(size_t)(cb + 16 + r) * NN + m] = g * acc1[j][r] + rb[(size_t)(cb + 16 + r) * NN + m];
        }
    }
}

// ---------------------------------------------------------------------------
extern "C" void kernel_launch(void* const* d_in, const int* in_sizes, int n_in,
                              void* d_out, int out_size, void* d_ws, size_t ws_size,
                              hipStream_t stream) {
    (void)in_sizes; (void)n_in; (void)out_size; (void)ws_size;
    const float* x    = (const float*)d_in[0];
    const float* y    = (const float*)d_in[1];
    const float* Wfx  = (const float*)d_in[2];
    const float* bfx  = (const float*)d_in[3];
    const float* Wgx  = (const float*)d_in[4];
    const float* bgx  = (const float*)d_in[5];
    const float* Whx  = (const float*)d_in[6];
    const float* bhx  = (const float*)d_in[7];
    const float* Wfy  = (const float*)d_in[8];
    const float* bfy  = (const float*)d_in[9];
    const float* Wgy  = (const float*)d_in[10];
    const float* bgy  = (const float*)d_in[11];
    const float* Why  = (const float*)d_in[12];
    const float* bhy  = (const float*)d_in[13];
    const float* gam  = (const float*)d_in[14];
    float* out_x = (float*)d_out;
    float* out_y = out_x + (size_t)BB * CC * NN;

    char* w = (char*)d_ws;
    auto alloc = [&](size_t bytes) { char* p = w; w += (bytes + 255) & ~(size_t)255; return p; };
    unsigned short* XT  = (unsigned short*)alloc((size_t)BB * NN * CC * 2);
    unsigned short* YT  = (unsigned short*)alloc((size_t)BB * NN * CC * 2);
    unsigned short* FXT = (unsigned short*)alloc((size_t)BB * NN * KC8 * 2);
    unsigned short* GXT = (unsigned short*)alloc((size_t)BB * NN * KC8 * 2);
    unsigned short* FYT = (unsigned short*)alloc((size_t)BB * NN * KC8 * 2);
    unsigned short* GYT = (unsigned short*)alloc((size_t)BB * NN * KC8 * 2);
    unsigned short* HX  = (unsigned short*)alloc((size_t)BB * CC * NN * 2);
    unsigned short* HY  = (unsigned short*)alloc((size_t)BB * CC * NN * 2);
    unsigned short* Wpx = (unsigned short*)alloc(320 * CC * 2);
    unsigned short* Wpy = (unsigned short*)alloc(320 * CC * 2);
    float* bpx = (float*)alloc(320 * 4);
    float* bpy = (float*)alloc(320 * 4);
    float* Lx  = (float*)alloc((size_t)BB * NN * 4);
    float* Ly  = (float*)alloc((size_t)BB * NN * 4);

    dim3 tb(32, 8);
    dim3 tg(NN / 32, CC / 32, BB);
    k_transpose_bf16<<<tg, tb, 0, stream>>>(x, XT);
    k_transpose_bf16<<<tg, tb, 0, stream>>>(y, YT);

    k_pack_w<<<320, CC, 0, stream>>>(Wfx, bfx, Wgx, bgx, Whx, bhx, Wpx, bpx);
    k_pack_w<<<320, CC, 0, stream>>>(Wfy, bfy, Wgy, bgy, Why, bhy, Wpy, bpy);

    dim3 pg(20 * (NN / 16) / 8, BB);
    k_proj<<<pg, 256, 0, stream>>>(XT, Wpx, bpx, FXT, GXT, HX);
    k_proj<<<pg, 256, 0, stream>>>(YT, Wpy, bpy, FYT, GYT, HY);

    k_passA<<<BB * (NN / 32) / 8, 256, 0, stream>>>(FYT, GXT, Lx);  // att_x rows
    k_passA<<<BB * (NN / 32) / 8, 256, 0, stream>>>(FXT, GYT, Ly);  // att_y rows

    dim3 bg2(NN / (16 * MT), BB);
    k_passB<<<bg2, 256, 0, stream>>>(FYT, GXT, HX, Lx, x, gam, out_x);
    k_passB<<<bg2, 256, 0, stream>>>(FXT, GYT, HY, Ly, y, gam, out_y);
}
